// Model_39676907880954
// MI455X (gfx1250) — matrix operand order repaired and checked by reading, not compiled, not run on hardware
//
#include <hip/hip_runtime.h>
#include <stdint.h>

#define D_MODEL 1024
#define FF_DIM  1024
#define BATCH   8
#define SEQ_T   2048
#define M_ROWS  (BATCH * SEQ_T)
#define SCALE_F 0.03125f

typedef __attribute__((ext_vector_type(16))) __bf16          v16bf;
typedef __attribute__((ext_vector_type(8)))  float           v8f;
typedef __attribute__((ext_vector_type(8)))  unsigned short  v8u;
typedef __attribute__((ext_vector_type(16))) unsigned short  v16u;

// ---------- helpers ----------

__device__ __forceinline__ unsigned short f2bf(float f) {
    union { float f; unsigned u; } v; v.f = f;
    unsigned r = v.u + 0x7FFFu + ((v.u >> 16) & 1u);   // round-to-nearest-even
    return (unsigned short)(r >> 16);
}

__device__ __forceinline__ v8f zero8f() {
    v8f z;
#pragma unroll
    for (int i = 0; i < 8; ++i) z[i] = 0.0f;
    return z;
}

// A fragment (16x32 bf16, MxK). ISA layout: lanes 0-15 (h=0): row m, K={0..7,16..23};
// lanes 16-31 (h=1): row m, K={8..15,24..31}. Two 16B contiguous loads per lane.
__device__ __forceinline__ v16bf load_a_bf16(const unsigned short* p, int ld, int lane) {
    const int m = lane & 15, h = lane >> 4;
    const unsigned short* r0 = p + m * ld + h * 8;
    v8u lo = *(const v8u*)(r0);
    v8u hi = *(const v8u*)(r0 + 16);
    v16u t;
#pragma unroll
    for (int i = 0; i < 8; ++i) { t[i] = lo[i]; t[8 + i] = hi[i]; }
    return __builtin_bit_cast(v16bf, t);
}

// A fragment from an f32 tile (converted to bf16 on the fly) - used for P (post-softmax).
__device__ __forceinline__ v16bf load_a_f32(const float* p, int ld, int lane) {
    const int m = lane & 15, h = lane >> 4;
    const float* r0 = p + m * ld + h * 8;
    v16u t;
#pragma unroll
    for (int i = 0; i < 8; ++i) { t[i] = f2bf(r0[i]); t[8 + i] = f2bf(r0[16 + i]); }
    return __builtin_bit_cast(v16bf, t);
}

// B fragment (32x16 bf16, KxN). Lane n<16: column n, K={0..7,16..23};
// lane n+16: column n, K={8..15,24..31}. Two 16B contiguous loads per lane.
// "column of B" must be contiguous in memory (row of W, row of K, row of V^T).
__device__ __forceinline__ v16bf load_b_bf16(const unsigned short* p, int ld, int lane) {
    const int n = lane & 15, h = lane >> 4;
    const unsigned short* r0 = p + n * ld + h * 8;
    v8u lo = *(const v8u*)(r0);
    v8u hi = *(const v8u*)(r0 + 16);
    v16u t;
#pragma unroll
    for (int i = 0; i < 8; ++i) { t[i] = lo[i]; t[8 + i] = hi[i]; }
    return __builtin_bit_cast(v16bf, t);
}

__device__ __forceinline__ v8f wmma_bf16(v16bf a, v16bf b, v8f c) {
    return __builtin_amdgcn_wmma_f32_16x16x32_bf16(
        /*neg_a=*/false, a, /*neg_b=*/false, b,
        /*c_mod=*/(short)0, c, /*reuse_a=*/false, /*reuse_b=*/false);
}

// ---------- kernels ----------

__global__ void cvt_kernel(const float* __restrict__ src,
                           unsigned short* __restrict__ dst, int n) {
    int i = blockIdx.x * blockDim.x + threadIdx.x;
    int stride = gridDim.x * blockDim.x;
    for (; i < n; i += stride) dst[i] = f2bf(src[i]);
}

// Y[m,f] = sum_k X[m,k] * W[f,k] + bias[f]   (X: [M_ROWS, D] bf16, W: [F, D] bf16)
// transposeOut=0 -> Y row-major [M_ROWS, F] bf16
// transposeOut=1 -> Y = V^T layout [BATCH][F][SEQ_T] bf16
__global__ void __launch_bounds__(256)
proj_kernel(const unsigned short* __restrict__ X,
            const unsigned short* __restrict__ W,
            const float* __restrict__ bias,
            unsigned short* __restrict__ Y, int transposeOut) {
    const int tid  = threadIdx.x;
    const int lane = tid & 31, wave = tid >> 5;
    const int wm = wave & 3, wn = wave >> 2;            // 4 waves in M x 2 in N
    const int rowBase = blockIdx.x * 128 + wm * 32;
    const int colBase = blockIdx.y * 128 + wn * 64;

    v8f acc[2][4];
#pragma unroll
    for (int i = 0; i < 2; ++i)
#pragma unroll
        for (int j = 0; j < 4; ++j) acc[i][j] = zero8f();

    for (int kk = 0; kk < D_MODEL; kk += 32) {
        v16bf a0 = load_a_bf16(X + rowBase * D_MODEL + kk, D_MODEL, lane);
        v16bf a1 = load_a_bf16(X + (rowBase + 16) * D_MODEL + kk, D_MODEL, lane);
#pragma unroll
        for (int j = 0; j < 4; ++j) {
            v16bf bj = load_b_bf16(W + (colBase + j * 16) * D_MODEL + kk, D_MODEL, lane);
            acc[0][j] = wmma_bf16(a0, bj, acc[0][j]);
            acc[1][j] = wmma_bf16(a1, bj, acc[1][j]);
        }
    }

    const int h = lane >> 4, n = lane & 15;
#pragma unroll
    for (int i = 0; i < 2; ++i) {
        const int r0 = rowBase + i * 16 + h * 8;
#pragma unroll
        for (int j = 0; j < 4; ++j) {
            const int col = colBase + j * 16 + n;
            const float bv = bias[col];
#pragma unroll
            for (int r = 0; r < 8; ++r) {
                const float v = acc[i][j][r] + bv;
                const int row = r0 + r;
                if (transposeOut) {
                    const int b = row >> 11;          // row / SEQ_T
                    const int t = row & (SEQ_T - 1);
                    Y[(b * FF_DIM + col) * SEQ_T + t] = f2bf(v);
                } else {
                    Y[row * FF_DIM + col] = f2bf(v);
                }
            }
        }
    }
}

// Flash attention: one block = (batch b, 32 query rows). 8 waves:
//  score phase: wave (mi,ni) computes S[mi*16.., ni*16..] over full K=1024
//  PV phase:    wave owns features [wave*128, wave*128+128)
__global__ void __launch_bounds__(256)
attn_kernel(const unsigned short* __restrict__ Q,
            const unsigned short* __restrict__ K,
            const unsigned short* __restrict__ Vt,   // [BATCH][FF_DIM][SEQ_T]
            float* __restrict__ O) {
    __shared__ unsigned short sQ[32][D_MODEL + 8];   // bf16 Q tile, padded stride
    __shared__ float sS[32][68];                     // score / P tile, padded
    __shared__ float sRow[32];                       // per-row corr then 1/sum

    const int tid  = threadIdx.x;
    const int lane = tid & 31, wave = tid >> 5;
    const int b     = blockIdx.y;
    const int qbase = blockIdx.x * 32;
    const int mi = wave & 1;        // score sub-tile row (0..1)
    const int ni = wave >> 1;       // score sub-tile col (0..3)
    const int fbase = wave * 128;   // this wave's feature slice for O

    // stage Q tile (32 x 1024 bf16) into LDS, vectorized 8 elems / op
    for (int c = tid; c < 32 * 128; c += 256) {
        const int row = c >> 7, col = (c & 127) * 8;
        *(v8u*)&sQ[row][col] =
            *(const v8u*)&Q[(b * SEQ_T + qbase + row) * FF_DIM + col];
    }
    __syncthreads();

    v8f o[2][8];
#pragma unroll
    for (int i = 0; i < 2; ++i)
#pragma unroll
        for (int j = 0; j < 8; ++j) o[i][j] = zero8f();

    float rmax = -3.0e38f, rsum = 0.0f;   // meaningful in wave 0 only

    for (int kt = 0; kt < SEQ_T; kt += 64) {
        // ---- scores: S(sub) = Q(mi) . K(kt+ni*16..)^T over K-dim 1024 ----
        v8f s = zero8f();
        const unsigned short* kptr = K + (b * SEQ_T + kt + ni * 16) * FF_DIM;
        if (kt + 64 < SEQ_T)   // prefetch next key tile (global_prefetch_b8)
            __builtin_prefetch(K + (b * SEQ_T + kt + 64 + ni * 16) * FF_DIM + lane * 64, 0, 0);
#pragma unroll 4
        for (int kk = 0; kk < FF_DIM; kk += 32) {
            v16bf a  = load_a_bf16(&sQ[mi * 16][0] + kk, D_MODEL + 8, lane);
            v16bf bb = load_b_bf16(kptr + kk, FF_DIM, lane);
            s = wmma_bf16(a, bb, s);
        }
        __syncthreads();   // prior-iteration P consumers are done with sS
        {
            const int h = lane >> 4, n = lane & 15;
            const int r0 = mi * 16 + h * 8;
#pragma unroll
            for (int r = 0; r < 8; ++r) sS[r0 + r][ni * 16 + n] = s[r] * SCALE_F;
        }
        __syncthreads();

        // ---- online softmax over this 32x64 tile (wave 0, one lane per row) ----
        if (wave == 0) {
            float mx = rmax;
#pragma unroll 8
            for (int c = 0; c < 64; ++c) mx = fmaxf(mx, sS[lane][c]);
            const float corr = __expf(rmax - mx);
            rmax = mx;
            float sum = 0.0f;
#pragma unroll 8
            for (int c = 0; c < 64; ++c) {
                const float p = __expf(sS[lane][c] - mx);
                sS[lane][c] = p;
                sum += p;
            }
            rsum = rsum * corr + sum;
            sRow[lane] = corr;
        }
        __syncthreads();

        // ---- rescale O accumulators by per-row correction ----
        {
            const int h = lane >> 4;
#pragma unroll
            for (int mt = 0; mt < 2; ++mt)
#pragma unroll
                for (int r = 0; r < 8; ++r) {
                    const float cr = sRow[mt * 16 + h * 8 + r];
#pragma unroll
                    for (int j = 0; j < 8; ++j) o[mt][j][r] *= cr;
                }
        }

        // ---- O += P @ V  (this wave's 128-feature slice; V^T rows contiguous) ----
#pragma unroll
        for (int ks = 0; ks < 2; ++ks) {
            v16bf p0 = load_a_f32(&sS[0][ks * 32], 68, lane);
            v16bf p1 = load_a_f32(&sS[16][ks * 32], 68, lane);
            const unsigned short* vptr =
                Vt + (b * FF_DIM + fbase) * SEQ_T + kt + ks * 32;
#pragma unroll
            for (int j = 0; j < 8; ++j) {
                v16bf vb = load_b_bf16(vptr + j * 16 * SEQ_T, SEQ_T, lane);
                o[0][j] = wmma_bf16(p0, vb, o[0][j]);
                o[1][j] = wmma_bf16(p1, vb, o[1][j]);
            }
        }
    }

    __syncthreads();
    if (wave == 0) sRow[lane] = 1.0f / rsum;
    __syncthreads();

    {
        const int h = lane >> 4, n = lane & 15;
#pragma unroll
        for (int mt = 0; mt < 2; ++mt)
#pragma unroll
            for (int r = 0; r < 8; ++r) {
                const int row = mt * 16 + h * 8 + r;
                const float inv = sRow[row];
                float* op = O + (b * SEQ_T + qbase + row) * FF_DIM + fbase + n;
#pragma unroll
                for (int j = 0; j < 8; ++j) op[j * 16] = o[mt][j][r] * inv;
            }
    }
}

// ---------- host launcher ----------

extern "C" void kernel_launch(void* const* d_in, const int* in_sizes, int n_in,
                              void* d_out, int out_size, void* d_ws, size_t ws_size,
                              hipStream_t stream) {
    (void)in_sizes; (void)n_in; (void)out_size; (void)ws_size;

    const float* seq = (const float*)d_in[0];
    const float* Wq  = (const float*)d_in[1];
    const float* bq  = (const float*)d_in[2];
    const float* Wk  = (const float*)d_in[3];
    const float* bk  = (const float*)d_in[4];
    const float* Wv  = (const float*)d_in[5];
    const float* bv  = (const float*)d_in[6];
    float* out = (float*)d_out;

    unsigned short* p = (unsigned short*)d_ws;
    unsigned short* seqb = p; p += (size_t)M_ROWS * D_MODEL;   // 32 MB
    unsigned short* wqb  = p; p += (size_t)FF_DIM * D_MODEL;   //  2 MB
    unsigned short* wkb  = p; p += (size_t)FF_DIM * D_MODEL;
    unsigned short* wvb  = p; p += (size_t)FF_DIM * D_MODEL;
    unsigned short* Qb   = p; p += (size_t)M_ROWS * FF_DIM;    // 32 MB
    unsigned short* Kb   = p; p += (size_t)M_ROWS * FF_DIM;
    unsigned short* Vtb  = p;                                  // 32 MB, [B][F][T]

    cvt_kernel<<<2048, 256, 0, stream>>>(seq, seqb, M_ROWS * D_MODEL);
    cvt_kernel<<<512, 256, 0, stream>>>(Wq, wqb, FF_DIM * D_MODEL);
    cvt_kernel<<<512, 256, 0, stream>>>(Wk, wkb, FF_DIM * D_MODEL);
    cvt_kernel<<<512, 256, 0, stream>>>(Wv, wvb, FF_DIM * D_MODEL);

    dim3 pgrid(M_ROWS / 128, FF_DIM / 128);   // 128 x 8
    proj_kernel<<<pgrid, 256, 0, stream>>>(seqb, wqb, bq, Qb, 0);
    proj_kernel<<<pgrid, 256, 0, stream>>>(seqb, wkb, bk, Kb, 0);
    proj_kernel<<<pgrid, 256, 0, stream>>>(seqb, wvb, bv, Vtb, 1);

    dim3 agrid(SEQ_T / 32, BATCH);            // 64 x 8
    attn_kernel<<<agrid, 256, 0, stream>>>(Qb, Kb, Vtb, out);
}
